// Model_74225624810082
// MI455X (gfx1250) — matrix operand order repaired and checked by reading, not compiled, not run on hardware
//
#include <hip/hip_runtime.h>
#include <hip/hip_bf16.h>
#include <math.h>

// Problem dims
#define B_   256
#define T_   512
#define F_   64
#define H_   128
#define G4_  512   // 4H
#define KG_  256   // 2F + H (fused gates K)

typedef __attribute__((ext_vector_type(16))) _Float16 v16h;
typedef __attribute__((ext_vector_type(8)))  _Float16 v8h;
typedef __attribute__((ext_vector_type(8)))  float    v8f;

// ---------------- workspace layout (bytes) ----------------
#define WS_WG    0          // gates weight packed  [256x512] f16 -> 262144
#define WS_WDH   262144     // W_dh packed [64x128]  f16 -> 16384
#define WS_WHR   278528     // W_hr packed [128x64]  f16 -> 16384
#define WS_WFR   294912     // W_fr packed [64x64]   f16 -> 8192
#define WS_WWC   303104     // W_wc packed [128x64]  f16 -> 16384
#define WS_BG    319488     // b_ih+b_hh   [512]     f32 -> 2048
#define WS_MSUM  321536     // msum[t]+eps [512]     f32 -> 2048
#define WS_ACC   323584     // {xloss, yloss_num, yloss_den, pad} f32

// ---------------- fragment loaders ----------------
// A fragment (16x32, f16) from LDS. Layout per ISA 7.12.2:
// lane<16: M=lane, halves 0..7 -> K+0..7, halves 8..15 -> K+16..23
// lane>=16: M=lane-16, halves 0..7 -> K+8..15, halves 8..15 -> K+24..31
__device__ inline v16h load_a_frag(const _Float16* sm, int ldk, int kcol0, int lane) {
  const int M  = lane & 15;
  const int hi = lane >> 4;
  const _Float16* p = sm + M * ldk + kcol0 + hi * 8;
  v8h lo = *(const v8h*)p;
  v8h hh = *(const v8h*)(p + 16);
  v16h a;
#pragma unroll
  for (int i = 0; i < 8; ++i) { a[i] = lo[i]; a[i + 8] = hh[i]; }
  return a;
}

// B fragment (32x16, f16) from a pre-swizzled weight: each lane's 16 halves
// are contiguous -> two 16B global loads.
__device__ inline v16h load_b_frag(const _Float16* wp, int nTiles, int kb, int nt, int lane) {
  const _Float16* p = wp + ((((kb * nTiles + nt) * 32) + lane) << 4);
  v8h lo = *(const v8h*)p;
  v8h hh = *(const v8h*)(p + 8);
  v16h b;
#pragma unroll
  for (int i = 0; i < 8; ++i) { b[i] = lo[i]; b[i + 8] = hh[i]; }
  return b;
}

__device__ inline v8f wmma16(v16h a, v16h b, v8f c) {
  return __builtin_amdgcn_wmma_f32_16x16x32_f16(false, a, false, b, (short)0, c, false, false);
}

__device__ inline float sigf(float x) { return 1.f / (1.f + __expf(-x)); }

// ---------------- setup kernels ----------------
// Generic pack: W is [N][K] row-major (out = A @ W^T -> B[k][n] = W[n][k]).
// dst element linear index = ((kb*nTiles + nt)*32 + lane)*16 + hh
__global__ void pack_b_kernel(const float* __restrict__ W, _Float16* __restrict__ dst,
                              int K, int N, int zero_diag) {
  int idx = blockIdx.x * blockDim.x + threadIdx.x;
  if (idx >= K * N) return;
  int hh   = idx & 15;
  int lane = (idx >> 4) & 31;
  int rest = idx >> 9;
  int nT = N >> 4;
  int nt = rest % nT;
  int kb = rest / nT;
  int k = kb * 32 + ((lane < 16) ? 0 : 8) + (hh & 7) + ((hh >> 3) << 4);
  int n = nt * 16 + (lane & 15);
  float v = W[n * K + k];
  if (zero_diag && k == n) v = 0.f;
  dst[idx] = (_Float16)v;
}

// Fused gates weight: K 0..127 -> W_ih[n][k], K 128..255 -> W_hh[n][k-128]
__global__ void pack_gates_kernel(const float* __restrict__ Wih, const float* __restrict__ Whh,
                                  _Float16* __restrict__ dst) {
  int idx = blockIdx.x * blockDim.x + threadIdx.x;
  if (idx >= KG_ * G4_) return;
  int hh   = idx & 15;
  int lane = (idx >> 4) & 31;
  int rest = idx >> 9;
  int nT = G4_ >> 4;   // 32
  int nt = rest % nT;
  int kb = rest / nT;
  int k = kb * 32 + ((lane < 16) ? 0 : 8) + (hh & 7) + ((hh >> 3) << 4);
  int n = nt * 16 + (lane & 15);
  float v = (k < 128) ? Wih[n * 128 + k] : Whh[n * 128 + (k - 128)];
  dst[idx] = (_Float16)v;
}

__global__ void setup_misc_kernel(const float* __restrict__ b_ih, const float* __restrict__ b_hh,
                                  float* __restrict__ bg, float* __restrict__ accv) {
  int g = blockIdx.x * blockDim.x + threadIdx.x;
  if (g < G4_) bg[g] = b_ih[g] + b_hh[g];
  if (g < 4) accv[g] = 0.f;
}

// msum[t] = sum_{b,f} masks[b,t,f] + eps  (per-step loss normalizer)
__global__ void msum_kernel(const float* __restrict__ masks, float* __restrict__ msum) {
  __shared__ float red[256];
  int t = blockIdx.x;
  int b = threadIdx.x;
  const float* p = masks + ((size_t)b * T_ + t) * F_;
  float s = 0.f;
#pragma unroll 8
  for (int f = 0; f < F_; ++f) s += p[f];
  red[b] = s;
  __syncthreads();
  for (int st = 128; st > 0; st >>= 1) {
    if (b < st) red[b] += red[b + st];
    __syncthreads();
  }
  if (b == 0) msum[t] = red[0] + 1e-5f;
}

// ---------------- persistent scan kernel ----------------
// grid = 16 workgroups (16 batch rows each), block = 256 threads = 8 wave32.
__global__ __launch_bounds__(256) void brits_main(
    const float* __restrict__ values, const float* __restrict__ masks,
    const float* __restrict__ deltas,
    const float* __restrict__ labels, const float* __restrict__ is_train,
    const float* __restrict__ b_dh, const float* __restrict__ W_dx,
    const float* __restrict__ b_dx, const float* __restrict__ b_hr,
    const float* __restrict__ b_fr, const float* __restrict__ b_wc,
    const float* __restrict__ W_out, const float* __restrict__ b_out,
    const _Float16* __restrict__ Wdh_p, const _Float16* __restrict__ Whr_p,
    const _Float16* __restrict__ Wfr_p, const _Float16* __restrict__ Wwc_p,
    const _Float16* __restrict__ Wg_p,
    const float* __restrict__ bg, const float* __restrict__ msum,
    float* __restrict__ accg, float* __restrict__ out) {

  __shared__ float    sH [16 * 128];   // canonical h (f32)
  __shared__ _Float16 sAa[16 * 256];   // A staging buffer A
  __shared__ _Float16 sAb[16 * 128];   // A staging buffer B (decayed h, f16)
  __shared__ _Float16 sGx[16 * 64];    // gamma_x
  __shared__ _Float16 sXh[16 * 64];    // x_h
  __shared__ _Float16 sZh[16 * 64];    // z_h
  __shared__ _Float16 sAl[16 * 64];    // alpha
  __shared__ float    sRed[256];

  const int tid  = threadIdx.x;
  const int wave = tid >> 5;
  const int lane = tid & 31;
  const int lrow = lane & 15;
  const int hi   = lane >> 4;
  const int wg   = blockIdx.x;
  const int col128 = wave * 16 + lrow;          // column for width-128 D fragments
  const int colQ   = (wave & 3) * 16 + lrow;    // column for width-64 D fragments

  float* preds = out + 1;
  float* imp   = out + 1 + B_;                  // imputations: same layout as values

  // LSTM cell state lives in registers (fragment mapping: r = j + 8*hi, col = col128)
  float c_reg[8];
#pragma unroll
  for (int j = 0; j < 8; ++j) c_reg[j] = 0.f;
  float xloss = 0.f;

  // per-thread elementwise mapping for [16][64] phases: 4 contiguous cols
  const int e0  = tid * 4;
  const int er  = e0 >> 6;       // row 0..15
  const int en  = e0 & 63;       // col base (multiple of 4)
  const int gbE = wg * 16 + er;

  float wdiag[4], bdxv[4];
#pragma unroll
  for (int i = 0; i < 4; ++i) {
    wdiag[i] = W_dx[(en + i) * F_ + (en + i)];
    bdxv[i]  = b_dx[en + i];
  }
  const float bdh_c = b_dh[col128];
  const float bhr_c = b_hr[colQ];
  const float bfr_c = b_fr[colQ];
  const float bwc_c = b_wc[colQ];
  float bg_c[4];
#pragma unroll
  for (int q = 0; q < 4; ++q) bg_c[q] = bg[col128 + 128 * q];

  for (int i = tid; i < 16 * 128; i += 256) sH[i] = 0.f;
  __syncthreads();

  for (int t = 0; t < T_; ++t) {
    const float inv_ms = 1.f / msum[t];

    // ---- P1: stage d (f16) into sAa[:,0:64]; gamma_x = exp(-relu(d*w+b)) ----
    {
      const size_t base = ((size_t)gbE * T_ + t) * F_ + en;
#pragma unroll
      for (int i = 0; i < 4; ++i) {
        float dv = deltas[base + i];
        sAa[er * 256 + en + i] = (_Float16)dv;
        float gx = __expf(-fmaxf(dv * wdiag[i] + bdxv[i], 0.f));
        sGx[er * 64 + en + i] = (_Float16)gx;
      }
    }
    __syncthreads();

    // ---- P2: gamma_h GEMM [16,64]x[64,128]; h *= gamma_h; stage h (f16) ----
    {
      v8f acc = {};
#pragma unroll
      for (int kb = 0; kb < 2; ++kb) {
        v16h a = load_a_frag(sAa, 256, kb * 32, lane);
        v16h b = load_b_frag(Wdh_p, 8, kb, wave, lane);
        acc = wmma16(a, b, acc);
      }
#pragma unroll
      for (int j = 0; j < 8; ++j) {
        int r = j + 8 * hi;
        float gam = __expf(-fmaxf(acc[j] + bdh_c, 0.f));
        float hv = sH[r * 128 + col128] * gam;
        sH[r * 128 + col128]  = hv;
        sAb[r * 128 + col128] = (_Float16)hv;
      }
    }
    __syncthreads();

    // ---- P3: waves 0-3: x_h GEMM [16,128]x[128,64] (+loss1, x_c staging);
    //          waves 4-7: stage [gamma_x | m] into sAa[:,64:192] ----
    if (wave < 4) {
      v8f acc = {};
#pragma unroll
      for (int kb = 0; kb < 4; ++kb) {
        v16h a = load_a_frag(sAb, 128, kb * 32, lane);
        v16h b = load_b_frag(Whr_p, 4, kb, wave, lane);
        acc = wmma16(a, b, acc);
      }
#pragma unroll
      for (int j = 0; j < 8; ++j) {
        int r = j + 8 * hi;
        float xh = acc[j] + bhr_c;
        size_t g = ((size_t)(wg * 16 + r) * T_ + t) * F_ + colQ;
        float mv = masks[g], xv = values[g];
        xloss += fabsf(xv - xh) * mv * inv_ms;
        sXh[r * 64 + colQ] = (_Float16)xh;
        float xc = mv * xv + (1.f - mv) * xh;
        sAa[r * 256 + colQ] = (_Float16)xc;
      }
    } else {
      int lt = tid - 128;   // 0..127
#pragma unroll
      for (int i = 0; i < 16; ++i) {
        int e  = lt * 16 + i;   // 0..2047
        int r  = e >> 7;
        int cc = e & 127;
        _Float16 v;
        if (cc < 64) {
          v = sGx[r * 64 + cc];
        } else {
          size_t g = ((size_t)(wg * 16 + r) * T_ + t) * F_ + (cc - 64);
          v = (_Float16)masks[g];
        }
        sAa[r * 256 + 64 + cc] = v;
      }
    }
    __syncthreads();

    // ---- P4: waves 0-3: z_h GEMM [16,64]x[64,64] (+loss2);
    //          waves 4-7: alpha GEMM [16,128]x[128,64] ----
    if (wave < 4) {
      v8f acc = {};
#pragma unroll
      for (int kb = 0; kb < 2; ++kb) {
        v16h a = load_a_frag(sAa, 256, kb * 32, lane);
        v16h b = load_b_frag(Wfr_p, 4, kb, wave, lane);
        acc = wmma16(a, b, acc);
      }
#pragma unroll
      for (int j = 0; j < 8; ++j) {
        int r = j + 8 * hi;
        float zh = acc[j] + bfr_c;
        size_t g = ((size_t)(wg * 16 + r) * T_ + t) * F_ + colQ;
        float mv = masks[g], xv = values[g];
        xloss += fabsf(xv - zh) * mv * inv_ms;
        sZh[r * 64 + colQ] = (_Float16)zh;
      }
    } else {
      v8f acc = {};
#pragma unroll
      for (int kb = 0; kb < 4; ++kb) {
        v16h a = load_a_frag(sAa + 64, 256, kb * 32, lane);
        v16h b = load_b_frag(Wwc_p, 4, kb, wave - 4, lane);
        acc = wmma16(a, b, acc);
      }
#pragma unroll
      for (int j = 0; j < 8; ++j) {
        int r = j + 8 * hi;
        sAl[r * 64 + colQ] = (_Float16)(acc[j] + bwc_c);
      }
    }
    __syncthreads();

    // ---- P5: c_h, loss3, c_c; write imputation; stage [c_c | m] for gates ----
    {
      const size_t base = ((size_t)gbE * T_ + t) * F_ + en;
#pragma unroll
      for (int i = 0; i < 4; ++i) {
        int n = en + i;
        float al = (float)sAl[er * 64 + n];
        float zh = (float)sZh[er * 64 + n];
        float xh = (float)sXh[er * 64 + n];
        float ch = al * zh + (1.f - al) * xh;
        float mv = masks[base + i], xv = values[base + i];
        xloss += fabsf(xv - ch) * mv * inv_ms;
        float cc = mv * xv + (1.f - mv) * ch;
        imp[base + i] = cc;
        sAa[er * 256 + n]      = (_Float16)cc;
        sAa[er * 256 + 64 + n] = (_Float16)mv;
      }
    }
    __syncthreads();

    // ---- P6: gates GEMM [16,256]x[256,512]; wave w owns N-tiles {w,w+8,w+16,w+24}
    //          so i/f/g/o of hidden col128 land in this thread's accumulators ----
    {
      v8f a0 = {}, a1 = {}, a2 = {}, a3 = {};
#pragma unroll
      for (int kb = 0; kb < 8; ++kb) {
        v16h a = (kb < 4) ? load_a_frag(sAa, 256, kb * 32, lane)
                          : load_a_frag(sAb, 128, (kb - 4) * 32, lane);
        v16h b0 = load_b_frag(Wg_p, 32, kb, wave,      lane);
        a0 = wmma16(a, b0, a0);
        v16h b1 = load_b_frag(Wg_p, 32, kb, wave + 8,  lane);
        a1 = wmma16(a, b1, a1);
        v16h b2 = load_b_frag(Wg_p, 32, kb, wave + 16, lane);
        a2 = wmma16(a, b2, a2);
        v16h b3 = load_b_frag(Wg_p, 32, kb, wave + 24, lane);
        a3 = wmma16(a, b3, a3);
      }
#pragma unroll
      for (int j = 0; j < 8; ++j) {
        int r = j + 8 * hi;
        float iv = sigf(a0[j] + bg_c[0]);
        float fv = sigf(a1[j] + bg_c[1]);
        float gv = tanhf(a2[j] + bg_c[2]);
        float ov = sigf(a3[j] + bg_c[3]);
        float cn = fv * c_reg[j] + iv * gv;
        c_reg[j] = cn;
        sH[r * 128 + col128] = ov * tanhf(cn);
      }
    }
    __syncthreads();
  }

  // ---- epilogue: reduce x_loss partials; per-row logits / BCE partials ----
  sRed[tid] = xloss;
  __syncthreads();
  for (int s = 128; s > 0; s >>= 1) {
    if (tid < s) sRed[tid] += sRed[tid + s];
    __syncthreads();
  }
  if (tid == 0) atomicAdd(&accg[0], sRed[0]);

  if (tid < 16) {
    int b = wg * 16 + tid;
    float s = 0.f;
#pragma unroll 8
    for (int k = 0; k < 128; ++k) s += sH[tid * 128 + k] * W_out[k];
    float y = s + b_out[0];
    preds[b] = sigf(y);
    float lab = labels[b], tr = is_train[b];
    float mv = fmaxf(-y, 0.f);
    float yl = y - y * lab + mv + logf(__expf(-mv) + __expf(-y - mv));
    atomicAdd(&accg[1], yl * tr);
    atomicAdd(&accg[2], tr);
  }
}

__global__ void brits_finalize(const float* __restrict__ accg, float* __restrict__ out) {
  if (threadIdx.x == 0) {
    out[0] = accg[0] / (float)T_ + 0.3f * (accg[1] / (accg[2] + 1e-5f));
  }
}

// ---------------- launcher ----------------
extern "C" void kernel_launch(void* const* d_in, const int* in_sizes, int n_in,
                              void* d_out, int out_size, void* d_ws, size_t ws_size,
                              hipStream_t stream) {
  (void)in_sizes; (void)n_in; (void)out_size; (void)ws_size;
  const float* values   = (const float*)d_in[0];
  const float* masks    = (const float*)d_in[1];
  const float* deltas   = (const float*)d_in[2];
  const float* labels   = (const float*)d_in[3];
  const float* is_train = (const float*)d_in[4];
  const float* W_dh  = (const float*)d_in[5];
  const float* b_dh  = (const float*)d_in[6];
  const float* W_dx  = (const float*)d_in[7];
  const float* b_dx  = (const float*)d_in[8];
  const float* W_hr  = (const float*)d_in[9];
  const float* b_hr  = (const float*)d_in[10];
  const float* W_fr  = (const float*)d_in[11];
  const float* b_fr  = (const float*)d_in[12];
  const float* W_wc  = (const float*)d_in[13];
  const float* b_wc  = (const float*)d_in[14];
  const float* W_ih  = (const float*)d_in[15];
  const float* b_ih  = (const float*)d_in[16];
  const float* W_hh  = (const float*)d_in[17];
  const float* b_hh  = (const float*)d_in[18];
  const float* W_out = (const float*)d_in[19];
  const float* b_out = (const float*)d_in[20];

  char* ws = (char*)d_ws;
  _Float16* Wg_p  = (_Float16*)(ws + WS_WG);
  _Float16* Wdh_p = (_Float16*)(ws + WS_WDH);
  _Float16* Whr_p = (_Float16*)(ws + WS_WHR);
  _Float16* Wfr_p = (_Float16*)(ws + WS_WFR);
  _Float16* Wwc_p = (_Float16*)(ws + WS_WWC);
  float*    bg    = (float*)(ws + WS_BG);
  float*    msum  = (float*)(ws + WS_MSUM);
  float*    accg  = (float*)(ws + WS_ACC);

  setup_misc_kernel<<<2, 256, 0, stream>>>(b_ih, b_hh, bg, accg);
  msum_kernel<<<T_, 256, 0, stream>>>(masks, msum);
  pack_b_kernel<<<(64 * 128 + 255) / 256, 256, 0, stream>>>(W_dh, Wdh_p, 64, 128, 0);
  pack_b_kernel<<<(128 * 64 + 255) / 256, 256, 0, stream>>>(W_hr, Whr_p, 128, 64, 0);
  pack_b_kernel<<<(64 * 64 + 255) / 256, 256, 0, stream>>>(W_fr, Wfr_p, 64, 64, 1);
  pack_b_kernel<<<(128 * 64 + 255) / 256, 256, 0, stream>>>(W_wc, Wwc_p, 128, 64, 0);
  pack_gates_kernel<<<(KG_ * G4_ + 255) / 256, 256, 0, stream>>>(W_ih, W_hh, Wg_p);

  brits_main<<<B_ / 16, 256, 0, stream>>>(
      values, masks, deltas, labels, is_train,
      b_dh, W_dx, b_dx, b_hr, b_fr, b_wc, W_out, b_out,
      Wdh_p, Whr_p, Wfr_p, Wwc_p, Wg_p, bg, msum,
      accg, (float*)d_out);

  brits_finalize<<<1, 32, 0, stream>>>(accg, (float*)d_out);
}
